// EncoderLayer_695784702388
// MI455X (gfx1250) — matrix operand order repaired and checked by reading, not compiled, not run on hardware
//
#include <hip/hip_runtime.h>
#include <hip/hip_bf16.h>

// ---------------------------------------------------------------------------
// MI455X (gfx1250) encoder layer. Compute-bound (~172 GFLOP vs <10us of HBM
// traffic at 23.3 TB/s), so every GEMM-shaped op runs on v_wmma_f32_16x16x32_bf16.
// All GEMM shapes are compile-time -> templated epilogues, immediate-offset
// addressing, no runtime branches in hot paths. Attention row-sums are computed
// on the matrix pipe (WMMA vs ones-matrix) instead of lane shuffles.
// ---------------------------------------------------------------------------

#define DEV __device__ __forceinline__

typedef __attribute__((ext_vector_type(16))) __bf16 v16bf;
typedef __attribute__((ext_vector_type(8)))  __bf16 v8bf;
typedef __attribute__((ext_vector_type(4)))  __bf16 v4bf;
typedef __attribute__((ext_vector_type(8)))  float  v8f;

constexpr int BB = 2, TT = 2048, EE = 1024, HH = 16, DD = 64, FF = 4096;
constexpr int MM = BB * TT;           // 4096 rows of activations

DEV __bf16 f2bf(float f) {
  unsigned int u = __builtin_bit_cast(unsigned int, f);
  u = (u + 0x7FFFu + ((u >> 16) & 1u)) >> 16;          // RNE
  unsigned short us = (unsigned short)u;
  return __builtin_bit_cast(__bf16, us);
}
DEV v8f vzero() {
  v8f z;
  #pragma unroll
  for (int i = 0; i < 8; ++i) z[i] = 0.0f;
  return z;
}

// A fragment: 16x32 tile, lane = row (M), K split: lanes0-15 hold K 0-15,
// lanes16-31 hold K 16-31 (matches B fragment). Two 16B chunks per lane.
DEV v16bf load_a_frag(const __bf16* base, int ld) {
  const int lane = threadIdx.x & 31;
  const int row  = lane & 15;
  const int kg   = (lane >> 4) * 16;
  const __bf16* r = base + (size_t)row * ld;
  v8bf c0 = *reinterpret_cast<const v8bf*>(r + kg);
  v8bf c1 = *reinterpret_cast<const v8bf*>(r + kg + 8);
  v16bf out;
  #pragma unroll
  for (int i = 0; i < 8; ++i) { out[i] = c0[i]; out[i + 8] = c1[i]; }
  return out;
}

// B fragment: 32x16 tile of W (row-major [N,K] -> lane = N row of W),
// lanes0-15 hold K 0-15, lanes16-31 hold K 16-31 (contiguous 32B per lane).
DEV v16bf load_b_frag(const __bf16* base, int ld) {
  const int lane = threadIdx.x & 31;
  const int col  = lane & 15;
  const int k0   = (lane >> 4) * 16;
  return *reinterpret_cast<const v16bf*>(base + (size_t)col * ld + k0);
}

DEV v8f wmma_bf16(v16bf a, v16bf b, v8f c) {
  return __builtin_amdgcn_wmma_f32_16x16x32_bf16(false, a, false, b,
                                                 (short)0, c, false, false);
}

// ---------------------------------------------------------------------------
// fp32 -> bf16 cast, 4 elements per thread (counts here are multiples of 1024)
__global__ void cvt_kernel(const float* __restrict__ in, __bf16* __restrict__ out, int n4) {
  const int i = blockIdx.x * blockDim.x + threadIdx.x;
  if (i < n4) {
    const float4 v = reinterpret_cast<const float4*>(in)[i];
    v4bf o;
    o[0] = f2bf(v.x); o[1] = f2bf(v.y); o[2] = f2bf(v.z); o[3] = f2bf(v.w);
    reinterpret_cast<v4bf*>(out)[i] = o;
  }
}

// ---------------------------------------------------------------------------
// LayerNorm over E=1024, one block (256 thr) per row, fp32 in -> bf16 out
__global__ __launch_bounds__(256)
void ln_kernel(const float* __restrict__ x, const float* __restrict__ g,
               const float* __restrict__ b, __bf16* __restrict__ out) {
  const int row = blockIdx.x;
  const float4 v4 = reinterpret_cast<const float4*>(x + (size_t)row * EE)[threadIdx.x];
  float s  = v4.x + v4.y + v4.z + v4.w;
  float sq = v4.x * v4.x + v4.y * v4.y + v4.z * v4.z + v4.w * v4.w;
  #pragma unroll
  for (int m = 1; m < 32; m <<= 1) { s += __shfl_xor(s, m, 32); sq += __shfl_xor(sq, m, 32); }
  __shared__ float rs[8], rq[8];
  const int w = threadIdx.x >> 5;
  if ((threadIdx.x & 31) == 0) { rs[w] = s; rq[w] = sq; }
  __syncthreads();
  float ts = 0.f, tq = 0.f;
  #pragma unroll
  for (int i = 0; i < 8; ++i) { ts += rs[i]; tq += rq[i]; }
  const float mean = ts * (1.0f / EE);
  const float rstd = rsqrtf(tq * (1.0f / EE) - mean * mean + 1e-5f);
  const float4 gv = reinterpret_cast<const float4*>(g)[threadIdx.x];
  const float4 bv = reinterpret_cast<const float4*>(b)[threadIdx.x];
  const float xv[4] = {v4.x, v4.y, v4.z, v4.w};
  const float gg[4] = {gv.x, gv.y, gv.z, gv.w};
  const float bb2[4] = {bv.x, bv.y, bv.z, bv.w};
  v4bf o;
  #pragma unroll
  for (int i = 0; i < 4; ++i) o[i] = f2bf((xv[i] - mean) * rstd * gg[i] + bb2[i]);
  reinterpret_cast<v4bf*>(out + (size_t)row * EE)[threadIdx.x] = o;
}

// ---------------------------------------------------------------------------
// WMMA GEMM: out = A[M,K](bf16) * W[N,K]^T(bf16) [*scale][+bias][+resid]
// Block 256 thr = 8 waves; block tile 128x128, wave tile 64x32 (4x2 WMMA tiles).
// N, K and epilogue shape are compile-time -> immediate-offset epilogue.
template <int N, int K, bool HAS_BIAS, bool HAS_RES, bool OUT_F, bool OUT_B, bool DO_SCALE>
__global__ __launch_bounds__(256)
void gemm_kernel(const __bf16* __restrict__ A, const __bf16* __restrict__ W,
                 const float* __restrict__ bias, const float* __restrict__ resid,
                 float* __restrict__ outF, __bf16* __restrict__ outB, float scale) {
  const int w    = threadIdx.x >> 5;
  const int lane = threadIdx.x & 31;
  const int wm = w & 1, wn = w >> 1;
  const int rowBase = blockIdx.x * 128 + wm * 64;
  const int colBase = blockIdx.y * 128 + wn * 32;

  v8f acc[4][2];
  #pragma unroll
  for (int i = 0; i < 4; ++i)
    #pragma unroll
    for (int j = 0; j < 2; ++j) acc[i][j] = vzero();

  for (int kk = 0; kk < K; kk += 32) {
    v16bf af[4], bf[2];
    #pragma unroll
    for (int mi = 0; mi < 4; ++mi)
      af[mi] = load_a_frag(A + (size_t)(rowBase + mi * 16) * K + kk, K);
    #pragma unroll
    for (int ni = 0; ni < 2; ++ni)
      bf[ni] = load_b_frag(W + (size_t)(colBase + ni * 16) * K + kk, K);
    #pragma unroll
    for (int mi = 0; mi < 4; ++mi)
      #pragma unroll
      for (int ni = 0; ni < 2; ++ni)
        acc[mi][ni] = wmma_bf16(af[mi], bf[ni], acc[mi][ni]);
  }

  #pragma unroll
  for (int mi = 0; mi < 4; ++mi) {
    const int m0 = rowBase + mi * 16 + 8 * (lane >> 4);   // first of 8 rows
    #pragma unroll
    for (int ni = 0; ni < 2; ++ni) {
      const int n = colBase + ni * 16 + (lane & 15);
      const float bval = HAS_BIAS ? bias[n] : 0.0f;
      const size_t base = (size_t)m0 * N + n;
      #pragma unroll
      for (int v = 0; v < 8; ++v) {
        float val = acc[mi][ni][v];
        if constexpr (DO_SCALE) val *= scale;
        if constexpr (HAS_BIAS) val += bval;
        if constexpr (HAS_RES)  val += resid[base + (size_t)v * N];
        if constexpr (OUT_F)    outF[base + (size_t)v * N] = val;
        if constexpr (OUT_B)    outB[base + (size_t)v * N] = f2bf(val);
      }
    }
  }
}

// ---------------------------------------------------------------------------
// Fused GEGLU GEMM: u = h2*W1^T + b1, out = u[:, :FF] * gelu_exact(u[:, FF:])
// Dual accumulator sets for a-columns (n) and gate-columns (n+FF).
template <int K>
__global__ __launch_bounds__(256)
void gemm_geglu_kernel(const __bf16* __restrict__ A, const __bf16* __restrict__ W1,
                       const float* __restrict__ b1, __bf16* __restrict__ outB) {
  const int w    = threadIdx.x >> 5;
  const int lane = threadIdx.x & 31;
  const int wm = w & 1, wn = w >> 1;
  const int rowBase = blockIdx.x * 128 + wm * 64;
  const int colBase = blockIdx.y * 128 + wn * 32;

  v8f accA[4][2], accG[4][2];
  #pragma unroll
  for (int i = 0; i < 4; ++i)
    #pragma unroll
    for (int j = 0; j < 2; ++j) { accA[i][j] = vzero(); accG[i][j] = vzero(); }

  for (int kk = 0; kk < K; kk += 32) {
    v16bf af[4], bfA[2], bfG[2];
    #pragma unroll
    for (int mi = 0; mi < 4; ++mi)
      af[mi] = load_a_frag(A + (size_t)(rowBase + mi * 16) * K + kk, K);
    #pragma unroll
    for (int ni = 0; ni < 2; ++ni) {
      bfA[ni] = load_b_frag(W1 + (size_t)(colBase + ni * 16) * K + kk, K);
      bfG[ni] = load_b_frag(W1 + (size_t)(colBase + ni * 16 + FF) * K + kk, K);
    }
    #pragma unroll
    for (int mi = 0; mi < 4; ++mi)
      #pragma unroll
      for (int ni = 0; ni < 2; ++ni) {
        accA[mi][ni] = wmma_bf16(af[mi], bfA[ni], accA[mi][ni]);
        accG[mi][ni] = wmma_bf16(af[mi], bfG[ni], accG[mi][ni]);
      }
  }

  #pragma unroll
  for (int mi = 0; mi < 4; ++mi) {
    const int m0 = rowBase + mi * 16 + 8 * (lane >> 4);
    #pragma unroll
    for (int ni = 0; ni < 2; ++ni) {
      const int n = colBase + ni * 16 + (lane & 15);
      const float ba = b1[n];
      const float bg = b1[n + FF];
      const size_t base = (size_t)m0 * FF + n;
      #pragma unroll
      for (int v = 0; v < 8; ++v) {
        const float a = accA[mi][ni][v] + ba;
        const float g = accG[mi][ni][v] + bg;
        const float f = a * (0.5f * g * (1.0f + erff(g * 0.70710678118654752f)));
        outB[base + (size_t)v * FF] = f2bf(f);
      }
    }
  }
}

// ---------------------------------------------------------------------------
// v[B,T,H*D] -> vt[B,H,D,T] (so PV GEMM is row x row)
__global__ void vtrans_kernel(const __bf16* __restrict__ v, __bf16* __restrict__ vt) {
  const int idx = blockIdx.x * blockDim.x + threadIdx.x;   // 2^22 total
  const int t = idx & (TT - 1);
  const int d = (idx >> 11) & (DD - 1);
  const int h = (idx >> 17) & (HH - 1);
  const int b = idx >> 21;
  vt[idx] = v[((size_t)(b * TT + t)) * EE + h * DD + d];
}

// ---------------------------------------------------------------------------
// Flash-style attention. Grid = B*H*(T/64) blocks, 128 thr (4 waves).
// Each wave: 16 q-rows, streaming softmax over key blocks of 64.
// Row max: 4-way tile max + 4 lane-shuffle steps (ds_bpermute) per 64 keys.
// Row sum: extra WMMA accumulator against an all-ones B matrix -- every lane's
// column of that C tile holds the row sum in the same layout as O, so the
// running rescale and the final 1/sum need no shuffles at all.
// q is pre-scaled by 1/sqrt(D) in its projection epilogue.
__global__ __launch_bounds__(128)
void attn_kernel(const __bf16* __restrict__ q, const __bf16* __restrict__ k,
                 const __bf16* __restrict__ vt, __bf16* __restrict__ attn) {
  const int bid = blockIdx.x;
  const int qt = bid & 31;             // T/64 = 32 q tiles
  const int h  = (bid >> 5) & 15;
  const int b  = bid >> 9;
  const int w    = threadIdx.x >> 5;
  const int lane = threadIdx.x & 31;
  const int qRow = qt * 64 + w * 16;

  __shared__ __bf16 pT[4][16 * 64];    // per-wave P transpose staging (2KB/wave)

  const __bf16* qbase = q + ((size_t)(b * TT + qRow)) * EE + h * DD;
  const v16bf aQ0 = load_a_frag(qbase, EE);        // d 0..31
  const v16bf aQ1 = load_a_frag(qbase + 32, EE);   // d 32..63

  v16bf ones;
  #pragma unroll
  for (int i = 0; i < 16; ++i) ones[i] = f2bf(1.0f);

  v8f o[4], oS;
  #pragma unroll
  for (int t = 0; t < 4; ++t) o[t] = vzero();
  oS = vzero();
  float rm[8];
  #pragma unroll
  for (int v = 0; v < 8; ++v) rm[v] = -3.0e38f;

  for (int kb = 0; kb < TT; kb += 64) {
    const __bf16* kbase = k + ((size_t)(b * TT + kb)) * EE + h * DD;
    // S = Q * K^T : four 16x16 key tiles, each reduced over d in two 32-steps
    v8f s[4];
    #pragma unroll
    for (int c = 0; c < 4; ++c) {
      v16bf bKa = load_b_frag(kbase + (size_t)(c * 16) * EE, EE);
      v16bf bKb = load_b_frag(kbase + (size_t)(c * 16) * EE + 32, EE);
      s[c] = wmma_bf16(aQ0, bKa, vzero());
      s[c] = wmma_bf16(aQ1, bKb, s[c]);
    }

    // row max: tile-combine then 4 shuffle steps across the 16-lane half
    float nm[8], esc[8];
    #pragma unroll
    for (int v = 0; v < 8; ++v) {
      float mc = fmaxf(fmaxf(s[0][v], s[1][v]), fmaxf(s[2][v], s[3][v]));
      #pragma unroll
      for (int msk = 1; msk < 16; msk <<= 1) mc = fmaxf(mc, __shfl_xor(mc, msk, 32));
      nm[v] = fmaxf(rm[v], mc);
      esc[v] = __expf(rm[v] - nm[v]);
      rm[v] = nm[v];
    }
    // P = exp(S - m), staged to LDS in A-fragment-friendly row-major layout
    #pragma unroll
    for (int v = 0; v < 8; ++v) {
      const int r = v + 8 * (lane >> 4);
      #pragma unroll
      for (int c = 0; c < 4; ++c) {
        const float pv = __expf(s[c][v] - nm[v]);
        pT[w][r * 64 + c * 16 + (lane & 15)] = f2bf(pv);
      }
      #pragma unroll
      for (int t = 0; t < 4; ++t) o[t][v] *= esc[v];
      oS[v] *= esc[v];
    }

    // O += P * Vt^T ; row sums ride along as WMMA vs ones-matrix
    const v16bf aP0 = load_a_frag(&pT[w][0], 64);    // keys kb+0..31
    const v16bf aP1 = load_a_frag(&pT[w][32], 64);   // keys kb+32..63
    const __bf16* vtb = vt + (((size_t)(b * HH + h)) * DD) * TT + kb;
    #pragma unroll
    for (int t = 0; t < 4; ++t) {
      v16bf bV0 = load_b_frag(vtb + (size_t)(t * 16) * TT, TT);
      v16bf bV1 = load_b_frag(vtb + (size_t)(t * 16) * TT + 32, TT);
      o[t] = wmma_bf16(aP0, bV0, o[t]);
      o[t] = wmma_bf16(aP1, bV1, o[t]);
    }
    oS = wmma_bf16(aP0, ones, oS);
    oS = wmma_bf16(aP1, ones, oS);
  }

  #pragma unroll
  for (int v = 0; v < 8; ++v) {
    const float inv = 1.0f / oS[v];    // lane's own column == row sum
    #pragma unroll
    for (int t = 0; t < 4; ++t) o[t][v] *= inv;
  }
  #pragma unroll
  for (int t = 0; t < 4; ++t)
    #pragma unroll
    for (int v = 0; v < 8; ++v) {
      const int m = v + 8 * (lane >> 4);
      const int tg = qRow + m;
      const int d = t * 16 + (lane & 15);
      attn[((size_t)(b * TT + tg)) * EE + h * DD + d] = f2bf(o[t][v]);
    }
}

// ---------------------------------------------------------------------------
extern "C" void kernel_launch(void* const* d_in, const int* in_sizes, int n_in,
                              void* d_out, int out_size, void* d_ws, size_t ws_size,
                              hipStream_t stream) {
  (void)in_sizes; (void)n_in; (void)out_size; (void)ws_size;
  const float* src  = (const float*)d_in[0];
  const float* Wq   = (const float*)d_in[1];
  const float* Wk   = (const float*)d_in[2];
  const float* Wv   = (const float*)d_in[3];
  const float* Wo   = (const float*)d_in[4];
  const float* W1   = (const float*)d_in[5];
  const float* b1   = (const float*)d_in[6];
  const float* W2   = (const float*)d_in[7];
  const float* b2   = (const float*)d_in[8];
  const float* ln1w = (const float*)d_in[9];
  const float* ln1b = (const float*)d_in[10];
  const float* ln2w = (const float*)d_in[11];
  const float* ln2b = (const float*)d_in[12];
  float* out = (float*)d_out;

  char* p = (char*)d_ws;
  auto take = [&](size_t bytes) { void* r = (void*)p; p += (bytes + 255) & ~size_t(255); return r; };
  __bf16* wq_b = (__bf16*)take((size_t)EE * EE * 2);
  __bf16* wk_b = (__bf16*)take((size_t)EE * EE * 2);
  __bf16* wv_b = (__bf16*)take((size_t)EE * EE * 2);
  __bf16* wo_b = (__bf16*)take((size_t)EE * EE * 2);
  __bf16* w1_b = (__bf16*)take((size_t)2 * FF * EE * 2);
  __bf16* w2_b = (__bf16*)take((size_t)EE * FF * 2);
  __bf16* h_b  = (__bf16*)take((size_t)MM * EE * 2);
  __bf16* q_b  = (__bf16*)take((size_t)MM * EE * 2);
  __bf16* k_b  = (__bf16*)take((size_t)MM * EE * 2);
  __bf16* v_b  = (__bf16*)take((size_t)MM * EE * 2);
  __bf16* vt_b = (__bf16*)take((size_t)BB * HH * DD * TT * 2);
  __bf16* at_b = (__bf16*)take((size_t)MM * EE * 2);
  float*  x_f  = (float*) take((size_t)MM * EE * 4);
  __bf16* h2_b = (__bf16*)take((size_t)MM * EE * 2);
  __bf16* f_b  = (__bf16*)take((size_t)MM * FF * 2);

  auto cvt = [&](const float* in, __bf16* o, int n) {
    cvt_kernel<<<(n / 4 + 255) / 256, 256, 0, stream>>>(in, o, n / 4);
  };
  cvt(Wq, wq_b, EE * EE);
  cvt(Wk, wk_b, EE * EE);
  cvt(Wv, wv_b, EE * EE);
  cvt(Wo, wo_b, EE * EE);
  cvt(W1, w1_b, 2 * FF * EE);
  cvt(W2, w2_b, EE * FF);

  // --- pre-norm attention block ---
  ln_kernel<<<MM, 256, 0, stream>>>(src, ln1w, ln1b, h_b);

  dim3 gE(MM / 128, EE / 128);       // 32 x 8
  // q = h*Wq^T * (1/sqrt(D)) -> bf16
  gemm_kernel<EE, EE, false, false, false, true, true>
      <<<gE, 256, 0, stream>>>(h_b, wq_b, nullptr, nullptr, nullptr, q_b, 0.125f);
  // k, v -> bf16
  gemm_kernel<EE, EE, false, false, false, true, false>
      <<<gE, 256, 0, stream>>>(h_b, wk_b, nullptr, nullptr, nullptr, k_b, 1.0f);
  gemm_kernel<EE, EE, false, false, false, true, false>
      <<<gE, 256, 0, stream>>>(h_b, wv_b, nullptr, nullptr, nullptr, v_b, 1.0f);

  vtrans_kernel<<<(BB * HH * DD * TT) / 256, 256, 0, stream>>>(v_b, vt_b);
  attn_kernel<<<BB * HH * (TT / 64), 128, 0, stream>>>(q_b, k_b, vt_b, at_b);

  // x = src + attn @ Wo^T   (fp32 residual kept for LN2 + final residual)
  gemm_kernel<EE, EE, false, true, true, false, false>
      <<<gE, 256, 0, stream>>>(at_b, wo_b, nullptr, src, x_f, nullptr, 1.0f);

  // --- pre-norm GEGLU FFN block ---
  ln_kernel<<<MM, 256, 0, stream>>>(x_f, ln2w, ln2b, h2_b);

  dim3 gF(MM / 128, FF / 128);       // 32 x 32
  gemm_geglu_kernel<EE><<<gF, 256, 0, stream>>>(h2_b, w1_b, b1, f_b);

  // out = x + f @ W2^T + b2
  gemm_kernel<EE, FF, true, true, true, false, false>
      <<<gE, 256, 0, stream>>>(f_b, w2_b, b2, x_f, out, nullptr, 1.0f);
}
